// LSTM_70789650973282
// MI455X (gfx1250) — matrix operand order repaired and checked by reading, not compiled, not run on hardware
//
#include <hip/hip_runtime.h>
#include <math.h>

typedef unsigned short ushort_t;
typedef __attribute__((ext_vector_type(16))) __bf16        v16bf;
typedef __attribute__((ext_vector_type(8)))  float         v8f;
typedef __attribute__((ext_vector_type(4)))  unsigned int  v4u;
typedef __attribute__((ext_vector_type(4)))  int           v4i;

#define T_STEPS 512
#define BATCH   64
#define DIM     1024
#define HID     1024
#define NGATE4  4096     // 4 gates * HID columns
#define KB32    32       // 32-wide K blocks in K=1024
#define CHUNK   64       // timesteps of PreG buffered per persistent launch
#define NWG     32       // workgroups of the persistent recurrent kernel

#if __has_builtin(__builtin_amdgcn_global_load_async_to_lds_b128) && \
    __has_builtin(__builtin_amdgcn_s_wait_asynccnt)
#define USE_ASYNC_LDS 1
typedef __attribute__((address_space(1))) v4i as1_v4i;   // global-space int4
typedef __attribute__((address_space(3))) v4i as3_v4i;   // LDS-space int4
#else
#define USE_ASYNC_LDS 0
#endif

// ---------------------------------------------------------------- helpers
__device__ __forceinline__ ushort_t f2bf(float f) {
  union { float f; unsigned u; } x; x.f = f;
  unsigned r = x.u + 0x7FFFu + ((x.u >> 16) & 1u);   // round-to-nearest-even
  return (ushort_t)(r >> 16);
}

__device__ __forceinline__ v16bf ld_frag(const ushort_t* p0, const ushort_t* p1) {
  union { v4u u[2]; v16bf v; } t;
  t.u[0] = *(const v4u*)p0;
  t.u[1] = *(const v4u*)p1;
  return t.v;
}

__device__ __forceinline__ float sigf(float x) { return 1.f / (1.f + __expf(-x)); }

// --------- LDS staging: async-to-LDS when available, reg round-trip otherwise
__device__ __forceinline__ void stage_issue(const ushort_t* g, ushort_t* l, v4u& pend) {
#if USE_ASYNC_LDS
  __builtin_amdgcn_global_load_async_to_lds_b128((as1_v4i*)g, (as3_v4i*)l, 0, 0);
  (void)pend;
#else
  pend = *(const v4u*)g;
  (void)l;
#endif
}
__device__ __forceinline__ void stage_complete(ushort_t* l, v4u& pend) {
#if USE_ASYNC_LDS
  __builtin_amdgcn_s_wait_asynccnt(0);
  (void)l; (void)pend;
#else
  *(v4u*)l = pend;
#endif
}

// ---------------- shared 64x16 GEMM core: acc[4] += A(64x1024) * B-tile-strip
// A row-major bf16 (stride 1024) streamed via double-buffered LDS;
// B pre-packed per 32x16 tile (lane-major, 16 contiguous bf16 per lane).
__device__ __forceinline__ void gemm_64x16(const ushort_t* __restrict__ Aglob,
                                           const ushort_t* __restrict__ Bp, // + ntile*KB32*512
                                           ushort_t* Abuf,                  // [2][64*32]
                                           v8f acc[4], int tid, int lane) {
  const int r = tid >> 2, c8 = (tid & 3) << 3;       // this thread's 16B of A tile
  const ushort_t* gs = Aglob + (size_t)r * 1024 + c8;
  ushort_t* ls = Abuf + r * 32 + c8;
  const int mrow = lane & 15, khalf = lane >> 4;
  v4u pend;
  stage_issue(gs, ls, pend);          // K-block 0 -> buffer 0
  stage_complete(ls, pend);
  __syncthreads();
  for (int kb = 0; kb < KB32; ++kb) {
    ushort_t* cur = Abuf + (kb & 1) * (64 * 32);
    ushort_t* nls = Abuf + ((kb + 1) & 1) * (64 * 32) + r * 32 + c8;
    if (kb + 1 < KB32) stage_issue(gs + (kb + 1) * 32, nls, pend);  // overlap with compute
    const ushort_t* bp = Bp + (size_t)kb * 512 + (size_t)lane * 16;
    v16bf bfrag = ld_frag(bp, bp + 8);
    v16bf af[4];
#pragma unroll
    for (int mt = 0; mt < 4; ++mt) {
      const ushort_t* ap = cur + (mt * 16 + mrow) * 32 + khalf * 16;
      af[mt] = ld_frag(ap, ap + 8);
    }
#pragma unroll
    for (int mt = 0; mt < 4; ++mt)
      acc[mt] = __builtin_amdgcn_wmma_f32_16x16x32_bf16(false, af[mt], false, bfrag,
                                                        (short)0, acc[mt], false, false);
    if (kb + 1 < KB32) stage_complete(nls, pend);
    __syncthreads();
  }
}

// ---------------------------------------------------------------- x -> bf16
__global__ void cvt_x(const float* __restrict__ x, ushort_t* __restrict__ xbf, size_t n) {
  size_t id = (size_t)blockIdx.x * blockDim.x + threadIdx.x;
  if (id < n) xbf[id] = f2bf(x[id]);
}

// ------------------------------------------- pack W into WMMA-B tile order
__global__ void pack_weights(const float* __restrict__ WI, const float* __restrict__ WF,
                             const float* __restrict__ WO, const float* __restrict__ WG,
                             ushort_t* __restrict__ Wxp, ushort_t* __restrict__ Whp) {
  size_t id = (size_t)blockIdx.x * blockDim.x + threadIdx.x;
  const size_t per = (size_t)DIM * NGATE4;
  if (id >= 2 * per) return;
  int    sel = (int)(id / per);              // 0 = x-part rows, 1 = h-part rows
  size_t rem = id - (size_t)sel * per;
  int k  = (int)(rem / NGATE4);
  int n4 = (int)(rem % NGATE4);
  int gi = n4 >> 10, n = n4 & 1023;          // gate order: I,F,O,G
  const float* W = (gi == 0) ? WI : (gi == 1) ? WF : (gi == 2) ? WO : WG;
  int row = sel ? (DIM + k) : k;             // W is [D+H][H] row-major
  float v = W[(size_t)row * HID + n];
  size_t tile = (size_t)(n4 >> 4) * KB32 + (size_t)(k >> 5);
  int lane = (((k >> 4) & 1) << 4) | (n4 & 15);
  int j = k & 15;
  ushort_t* dst = sel ? Whp : Wxp;
  dst[tile * 512 + (size_t)lane * 16 + j] = f2bf(v);
}

// ------------------------------- bias concat + h0/c0 zeroing + barrier reset
__global__ void init_state(const float* __restrict__ bI, const float* __restrict__ bF,
                           const float* __restrict__ bO, const float* __restrict__ bG,
                           float* __restrict__ biasc, ushort_t* __restrict__ h0,
                           float* __restrict__ c0, unsigned* __restrict__ bar) {
  int id = blockIdx.x * blockDim.x + threadIdx.x;
  if (id < NGATE4) {
    int gi = id >> 10, n = id & 1023;
    const float* b = (gi == 0) ? bI : (gi == 1) ? bF : (gi == 2) ? bO : bG;
    biasc[id] = b[n];
  }
  if (id < BATCH * HID) { h0[id] = 0; c0[id] = 0.f; }
  if (id < 2) bar[id] = 0u;
}

// ---------------------------------------------------- input-projection GEMM
// Writes PreG in C-fragment layout: [lt][ntile(256)][mt(4)][lane(32)][8] f32.
// One 64-row block == one timestep's batch.
__global__ void __launch_bounds__(256) pregemm(const ushort_t* __restrict__ xbf_c,
                                               const ushort_t* __restrict__ Wxp,
                                               const float* __restrict__ biasc,
                                               float* __restrict__ pregp) {
  __shared__ ushort_t Abuf[2 * 64 * 32];
  int tid = threadIdx.x, wave = tid >> 5, lane = tid & 31;
  int lt    = blockIdx.x >> 5;                          // local timestep
  int ncol0 = ((blockIdx.x & 31) << 7) + (wave << 4);   // 16-col strip
  int ntile = ncol0 >> 4;
  v8f acc[4] = {};
  gemm_64x16(xbf_c + (size_t)lt * BATCH * DIM,
             Wxp + (size_t)ntile * KB32 * 512, Abuf, acc, tid, lane);
  float bias = biasc[ncol0 + (lane & 15)];
#pragma unroll
  for (int mt = 0; mt < 4; ++mt) {
    v8f v = acc[mt];
#pragma unroll
    for (int r = 0; r < 8; ++r) v[r] += bias;
    float* dst = pregp + ((((size_t)lt * 256 + ntile) * 4 + mt) * 32 + lane) * 8;
    union { v8f v; v4u u[2]; } t; t.v = v;
    *(v4u*)dst = t.u[0];
    *(v4u*)(dst + 4) = t.u[1];
  }
}

// ------------------------- persistent recurrent kernel: CHUNK steps, 32 WGs
// WG owns 32 hidden cols x 4 gates; grid barrier between timesteps.
__global__ void __launch_bounds__(256) lstm_chunk(
    const ushort_t* __restrict__ Whp, const float* __restrict__ pregp,
    float* __restrict__ cbuf, ushort_t* __restrict__ h0buf, ushort_t* __restrict__ h1buf,
    float* __restrict__ outc, float* __restrict__ htail, float* __restrict__ ctail,
    unsigned* __restrict__ bar, int c0, int nsteps, int last_chunk) {
  __shared__ ushort_t Abuf[2 * 64 * 32];
  __shared__ float Gbuf[4][BATCH][32];
  int tid = threadIdx.x, wave = tid >> 5, lane = tid & 31;
  int gi = wave >> 1, ct = wave & 1;
  int nh0 = blockIdx.x << 5;                    // first hidden col of this WG
  int n4base = (gi << 10) + nh0 + (ct << 4);    // gate-column strip base
  int ntile = n4base >> 4;
  int lcol = (ct << 4) + (lane & 15);
  int mbase = (lane >> 4) * 8;

  for (int lt = 0; lt < nsteps; ++lt) {
    const ushort_t* hprev = (lt & 1) ? h1buf : h0buf;
    ushort_t*       hnext = (lt & 1) ? h0buf : h1buf;
    const float* preg = pregp + (size_t)lt * (BATCH * (size_t)NGATE4);
    float* out_t = outc + (size_t)lt * (BATCH * HID);

    v8f acc[4] = {};
    gemm_64x16(hprev, Whp + (size_t)ntile * KB32 * 512, Abuf, acc, tid, lane);

    // add precomputed input projection (fragment-layout b128 loads) -> LDS
#pragma unroll
    for (int mt = 0; mt < 4; ++mt) {
      const float* sp = preg + (((size_t)ntile * 4 + mt) * 32 + lane) * 8;
      union { v4u u[2]; float f[8]; } t;
      t.u[0] = *(const v4u*)sp;
      t.u[1] = *(const v4u*)(sp + 4);
#pragma unroll
      for (int r = 0; r < 8; ++r)
        Gbuf[gi][mt * 16 + mbase + r][lcol] = acc[mt][r] + t.f[r];
    }
    __syncthreads();

    // fused cell update: 64x32 elems, 8 per thread
#pragma unroll
    for (int i = 0; i < 8; ++i) {
      int idx = tid * 8 + i;
      int m = idx >> 5, nl = idx & 31;
      int ng = nh0 + nl;
      float iv = sigf(Gbuf[0][m][nl]);
      float fv = sigf(Gbuf[1][m][nl]);
      float ov = sigf(Gbuf[2][m][nl]);
      float gv = tanhf(Gbuf[3][m][nl]);
      size_t ci = (size_t)m * HID + ng;
      float cn = fv * cbuf[ci] + iv * gv;
      float hv = ov * tanhf(cn);
      cbuf[ci]  = cn;
      hnext[ci] = f2bf(hv);
      out_t[ci] = hv;
      if (last_chunk && lt == nsteps - 1) { htail[ci] = hv; ctail[ci] = cn; }
    }

    // ---- device-scope grid barrier between timesteps ----
    __threadfence();            // release this thread's h/c/out stores
    __syncthreads();
    if (tid == 0) {
      unsigned target = (unsigned)(c0 + lt + 1);
      unsigned old = __hip_atomic_fetch_add(&bar[0], 1u, __ATOMIC_ACQ_REL,
                                            __HIP_MEMORY_SCOPE_AGENT);
      if (old == NWG - 1) {     // last to arrive: reset count, publish generation
        __hip_atomic_store(&bar[0], 0u, __ATOMIC_RELAXED, __HIP_MEMORY_SCOPE_AGENT);
        __hip_atomic_store(&bar[1], target, __ATOMIC_RELEASE, __HIP_MEMORY_SCOPE_AGENT);
      } else {
        while (__hip_atomic_load(&bar[1], __ATOMIC_ACQUIRE,
                                 __HIP_MEMORY_SCOPE_AGENT) < target)
          __builtin_amdgcn_s_sleep(2);
      }
      __threadfence();          // acquire: invalidate WGP caches
    }
    __syncthreads();
    asm volatile("" ::: "memory");
  }
}

// ------------------------------------------------------------------- launch
extern "C" void kernel_launch(void* const* d_in, const int* in_sizes, int n_in,
                              void* d_out, int out_size, void* d_ws, size_t ws_size,
                              hipStream_t stream) {
  (void)in_sizes; (void)n_in; (void)out_size; (void)ws_size;
  const float* x  = (const float*)d_in[0];
  const float* WI = (const float*)d_in[1];
  const float* bI = (const float*)d_in[2];
  const float* WF = (const float*)d_in[3];
  const float* bF = (const float*)d_in[4];
  const float* WO = (const float*)d_in[5];
  const float* bO = (const float*)d_in[6];
  const float* WG = (const float*)d_in[7];
  const float* bG = (const float*)d_in[8];
  float* out   = (float*)d_out;
  float* htail = out + (size_t)T_STEPS * BATCH * HID;
  float* ctail = htail + (size_t)BATCH * HID;

  char* w = (char*)d_ws;
  ushort_t* xbf   = (ushort_t*)(w);                  //  64 MiB: [T*B][D] bf16
  ushort_t* Wxp   = (ushort_t*)(w + 67108864);       //   8 MiB packed
  ushort_t* Whp   = (ushort_t*)(w + 75497472);       //   8 MiB packed
  float*    biasc = (float*)   (w + 83886080);       //  16 KiB
  float*    PreGp = (float*)   (w + 83902464);       //  64 MiB: [CHUNK][256][4][32][8] f32
  ushort_t* h0    = (ushort_t*)(w + 151011328);      // 128 KiB
  ushort_t* h1    = (ushort_t*)(w + 151142400);      // 128 KiB
  float*    cbuf  = (float*)   (w + 151273472);      // 256 KiB
  unsigned* bar   = (unsigned*)(w + 151535616);      // grid-barrier state
  // total ~145 MiB

  size_t nx = (size_t)T_STEPS * BATCH * DIM;
  cvt_x<<<(unsigned)((nx + 255) / 256), 256, 0, stream>>>(x, xbf, nx);
  size_t nw = 2 * (size_t)DIM * NGATE4;
  pack_weights<<<(unsigned)((nw + 255) / 256), 256, 0, stream>>>(WI, WF, WO, WG, Wxp, Whp);
  init_state<<<(BATCH * HID + 255) / 256, 256, 0, stream>>>(bI, bF, bO, bG, biasc, h0, cbuf, bar);

  for (int c0 = 0; c0 < T_STEPS; c0 += CHUNK) {
    pregemm<<<CHUNK * 32, 256, 0, stream>>>(
        xbf + (size_t)c0 * BATCH * DIM, Wxp, biasc, PreGp);
    lstm_chunk<<<NWG, 256, 0, stream>>>(
        Whp, PreGp, cbuf, h0, h1, out + (size_t)c0 * BATCH * HID,
        htail, ctail, bar, c0, CHUNK, (c0 + CHUNK >= T_STEPS) ? 1 : 0);
  }
}
